// ShiftedWindowAttention_90924457657009
// MI455X (gfx1250) — compile-verified, matrix operand order checked by reading, not run on hardware
//
#include <hip/hip_runtime.h>
#include <cstddef>

typedef _Float16 v8h  __attribute__((ext_vector_type(8)));
typedef _Float16 v16h __attribute__((ext_vector_type(16)));
typedef float    v8f  __attribute__((ext_vector_type(8)));

#define WINSZ 7
#define SHF   3
#define WSQ   49      // tokens per window
#define NHEAD 4
#define HDIM  32
#define CHN   128
#define IMGE  56

constexpr int XS_STRIDE  = 136;  // halves (also holds O after QKV stage)
constexpr int QKV_STRIDE = 392;  // halves
constexpr int S_STRIDE   = 52;   // floats
constexpr int P_STRIDE   = 72;   // halves

__device__ __forceinline__ v8f wmma_f16(v16h a, v16h b, v8f c) {
  return __builtin_amdgcn_wmma_f32_16x16x32_f16(false, a, false, b, (short)0, c,
                                                false, false);
}

// A-fragment (16x32 f16, M x K), row-major source. Also serves as the
// B-fragment of a transposed matrix (Kᵀ) since lane<->column mapping matches.
__device__ __forceinline__ v16h ldA_row(const _Float16* base, int stride,
                                        int row0, int k0, int lane) {
  const int l  = lane & 15;
  const int kk = k0 + ((lane & 16) ? 8 : 0);
  const _Float16* p = base + (size_t)(row0 + l) * stride + kk;
  v8h lo = *(const v8h*)(p);        // K = kk+0..7
  v8h hi = *(const v8h*)(p + 16);   // K = kk+16..23
  return __builtin_shufflevector(lo, hi, 0,1,2,3,4,5,6,7,8,9,10,11,12,13,14,15);
}

// B-fragment (32x16 f16, K x N) from a row-major K x N source (column gather).
__device__ __forceinline__ v16h ldB_col(const _Float16* base, int stride,
                                        int k0, int lane) {
  const int l  = lane & 15;
  const int kk = k0 + ((lane & 16) ? 8 : 0);
  const _Float16* p = base + (size_t)kk * stride + l;
  v16h b;
  #pragma unroll
  for (int i = 0; i < 8; ++i) {
    b[i]     = p[(size_t)i * stride];
    b[i + 8] = p[(size_t)(16 + i) * stride];
  }
  return b;
}

// B-fragment from pre-swizzled pack: 512 halves per fragment, 16/lane contiguous.
__device__ __forceinline__ v16h ldB_pack(const _Float16* frag, int lane) {
  const _Float16* p = frag + lane * 16;
  v8h lo = *(const v8h*)(p);
  v8h hi = *(const v8h*)(p + 8);
  return __builtin_shufflevector(lo, hi, 0,1,2,3,4,5,6,7,8,9,10,11,12,13,14,15);
}

__device__ __forceinline__ int region56(int g) {
  return g < (IMGE - WINSZ) ? 0 : (g < (IMGE - SHF) ? 1 : 2);
}

// ---------------------------------------------------------------------------
// Pre-pass: convert f32 weights to f16, swizzled into WMMA B-fragment layout.
// Fragment f = kc*(N/16)+nt holds 32 lanes x 16 halves.
// lane<16 : col n = nt*16+lane, halves = K {0..7, 16..23}+kc*32
// lane>=16: col n = nt*16+lane-16, halves = K {8..15,24..31}+kc*32
// ---------------------------------------------------------------------------
__global__ void pack_weights_kernel(const float* __restrict__ W,
                                    _Float16* __restrict__ P, int K, int N) {
  int idx = blockIdx.x * blockDim.x + threadIdx.x;
  if (idx >= K * N) return;
  int f    = idx >> 9;
  int lane = (idx >> 4) & 31;
  int i    = idx & 15;
  int NT   = N >> 4;
  int kc   = f / NT, nt = f - kc * NT;
  int l    = lane & 15;
  int hi   = (lane >> 4) & 1;
  int k    = kc * 32 + (i < 8 ? (hi ? 8 : 0) + i : (hi ? 24 : 16) + (i - 8));
  int n    = nt * 16 + l;
  P[idx] = (_Float16)W[(size_t)k * N + n];
}

// ---------------------------------------------------------------------------
// Fused shifted-window attention: one workgroup (8 waves) per window.
// ---------------------------------------------------------------------------
__global__ __launch_bounds__(256) void swin_window_attn_kernel(
    const float* __restrict__ x, const float* __restrict__ qkv_bias,
    const float* __restrict__ proj_bias, const float* __restrict__ rel_tab,
    const _Float16* __restrict__ qkv_pack, const _Float16* __restrict__ proj_pack,
    float* __restrict__ out) {
  __shared__ _Float16 lds_xs[64 * XS_STRIDE];    // X tile, later O tile
  __shared__ _Float16 lds_qkv[64 * QKV_STRIDE];  // Q|K|V (384 cols, q pre-scaled)
  __shared__ float    lds_s[64 * S_STRIDE];      // raw scores (per head)
  __shared__ _Float16 lds_p[64 * P_STRIDE];      // softmax probs (per head)
  __shared__ float    lds_rp[169 * NHEAD];       // rel-pos bias table

  const int tid  = threadIdx.x;
  const int lane = tid & 31;
  const int wv   = tid >> 5;  // 8 waves

  const int blk = blockIdx.x;           // 4096 = 64 imgs * 8 * 8 windows
  const int img = blk >> 6;
  const int wy  = (blk >> 3) & 7;
  const int wx  = blk & 7;

  // ---- Stage A: gather shifted window into LDS (f32 -> f16) ----------------
  for (int j = tid; j < 169 * NHEAD; j += 256) lds_rp[j] = rel_tab[j];

  for (int j = tid; j < WSQ * 32; j += 256) {
    int t  = j >> 5;
    int q  = (j & 31) << 2;
    int ty = t / WINSZ, tx = t - ty * WINSZ;
    int gr = wy * WINSZ + ty + SHF; if (gr >= IMGE) gr -= IMGE;
    int gc = wx * WINSZ + tx + SHF; if (gc >= IMGE) gc -= IMGE;
    const float4 v =
        *(const float4*)(x + (((size_t)img * IMGE + gr) * IMGE + gc) * CHN + q);
    _Float16* d = lds_xs + t * XS_STRIDE + q;
    d[0] = (_Float16)v.x; d[1] = (_Float16)v.y;
    d[2] = (_Float16)v.z; d[3] = (_Float16)v.w;
  }
  for (int j = tid; j < (64 - WSQ) * CHN; j += 256)   // zero-pad rows 49..63
    lds_xs[(WSQ + (j >> 7)) * XS_STRIDE + (j & 127)] = (_Float16)0;
  for (int j = tid; j < (64 - WSQ) * 64; j += 256)    // P rows 49..63 stay zero
    lds_p[(WSQ + (j >> 6)) * P_STRIDE + (j & 63)] = (_Float16)0;

  __syncthreads();

  // ---- Stage B: QKV = X @ Wqkv + b (q columns pre-scaled by 1/sqrt(32)) ----
  const float qscale = 0.17677669529663687f;
  for (int t = wv; t < 4 * 24; t += 8) {
    int mt = t / 24, nt = t - mt * 24;
    int n0 = nt * 16;
    float bv = qkv_bias[n0 + (lane & 15)];
    v8f acc;
    #pragma unroll
    for (int i = 0; i < 8; ++i) acc[i] = bv;
    #pragma unroll
    for (int kc = 0; kc < 4; ++kc) {
      v16h a = ldA_row(lds_xs, XS_STRIDE, mt * 16, kc * 32, lane);
      v16h b = ldB_pack(qkv_pack + (size_t)(kc * 24 + nt) * 512, lane);
      acc = wmma_f16(a, b, acc);
    }
    float sc = (n0 < CHN) ? qscale : 1.0f;
    int col   = n0 + (lane & 15);
    int rbase = mt * 16 + ((lane & 16) ? 8 : 0);
    #pragma unroll
    for (int r = 0; r < 8; ++r)
      lds_qkv[(rbase + r) * QKV_STRIDE + col] = (_Float16)(acc[r] * sc);
  }
  __syncthreads();

  // ---- per-head: scores -> softmax -> PV ----------------------------------
  const int rh_w = region56(wy * WINSZ);  // only wy==7 / wx==7 windows mix
  for (int h = 0; h < NHEAD; ++h) {
    // Stage C: S = Q Kᵀ (+rel-pos bias +shift mask). 16 tiles / 8 waves.
    for (int t = wv; t < 16; t += 8) {
      int mt = t >> 2, nt = t & 3;
      v16h a = ldA_row(lds_qkv, QKV_STRIDE, mt * 16, h * HDIM, lane);        // Q
      v16h b = ldA_row(lds_qkv, QKV_STRIDE, nt * 16, CHN + h * HDIM, lane);  // Kᵀ
      v8f acc = {};
      acc = wmma_f16(a, b, acc);

      int tc    = nt * 16 + (lane & 15);
      int rbase = mt * 16 + ((lane & 16) ? 8 : 0);
      if (tc < WSQ) {
        int ty  = tc / WINSZ, tx = tc - ty * WINSZ;
        int rht = region56(wy * WINSZ + ty);
        int rwt = region56(wx * WINSZ + tx);
        #pragma unroll
        for (int r = 0; r < 8; ++r) {
          int s = rbase + r;
          if (s < WSQ) {
            int sy = s / WINSZ, sx = s - sy * WINSZ;
            float bias =
                lds_rp[((sy - ty + 6) * 13 + (sx - tx + 6)) * NHEAD + h];
            bool same = (region56(wy * WINSZ + sy) == rht) &&
                        (region56(wx * WINSZ + sx) == rwt);
            lds_s[s * S_STRIDE + tc] = acc[r] + bias + (same ? 0.f : -100.f);
          }
        }
      }
    }
    __syncthreads();

    // Softmax over the 49 real columns; write P (f16), zero cols 49..63.
    if (tid < WSQ) {
      const float* row = lds_s + tid * S_STRIDE;
      float mx = -1e30f;
      for (int j = 0; j < WSQ; ++j) mx = fmaxf(mx, row[j]);
      float sum = 0.f;
      for (int j = 0; j < WSQ; ++j) sum += __expf(row[j] - mx);
      float inv = 1.0f / sum;
      _Float16* pr = lds_p + tid * P_STRIDE;
      for (int j = 0; j < WSQ; ++j)
        pr[j] = (_Float16)(__expf(row[j] - mx) * inv);
      for (int j = WSQ; j < 64; ++j) pr[j] = (_Float16)0;
    }
    __syncthreads();

    // Stage D: O[:, h*32 : h*32+32] = P @ V. 8 tiles, one per wave.
    {
      int mt = wv >> 1, nt = wv & 1;
      v8f acc = {};
      #pragma unroll
      for (int kc = 0; kc < 2; ++kc) {
        v16h a = ldA_row(lds_p, P_STRIDE, mt * 16, kc * 32, lane);
        v16h b = ldB_col(lds_qkv + 2 * CHN + h * HDIM + nt * 16, QKV_STRIDE,
                         kc * 32, lane);
        acc = wmma_f16(a, b, acc);
      }
      int col   = h * HDIM + nt * 16 + (lane & 15);
      int rbase = mt * 16 + ((lane & 16) ? 8 : 0);
      #pragma unroll
      for (int r = 0; r < 8; ++r)
        lds_xs[(rbase + r) * XS_STRIDE + col] = (_Float16)acc[r];
    }
    __syncthreads();
  }

  // ---- Stage E: Y = O @ Wproj + b, scatter with reverse shift -------------
  for (int t = wv; t < 32; t += 8) {
    int mt = t >> 3, nt = t & 7;
    int n0 = nt * 16;
    float bv = proj_bias[n0 + (lane & 15)];
    v8f acc;
    #pragma unroll
    for (int i = 0; i < 8; ++i) acc[i] = bv;
    #pragma unroll
    for (int kc = 0; kc < 4; ++kc) {
      v16h a = ldA_row(lds_xs, XS_STRIDE, mt * 16, kc * 32, lane);
      v16h b = ldB_pack(proj_pack + (size_t)(kc * 8 + nt) * 512, lane);
      acc = wmma_f16(a, b, acc);
    }
    int col   = n0 + (lane & 15);
    int rbase = mt * 16 + ((lane & 16) ? 8 : 0);
    #pragma unroll
    for (int r = 0; r < 8; ++r) {
      int s = rbase + r;
      if (s < WSQ) {
        int ty = s / WINSZ, tx = s - ty * WINSZ;
        int gr = wy * WINSZ + ty + SHF; if (gr >= IMGE) gr -= IMGE;
        int gc = wx * WINSZ + tx + SHF; if (gc >= IMGE) gc -= IMGE;
        out[(((size_t)img * IMGE + gr) * IMGE + gc) * CHN + col] = acc[r];
      }
    }
  }
}

extern "C" void kernel_launch(void* const* d_in, const int* in_sizes, int n_in,
                              void* d_out, int out_size, void* d_ws,
                              size_t ws_size, hipStream_t stream) {
  const float* x     = (const float*)d_in[0];
  const float* qkvW  = (const float*)d_in[1];
  const float* qkvB  = (const float*)d_in[2];
  const float* projW = (const float*)d_in[3];
  const float* projB = (const float*)d_in[4];
  const float* relT  = (const float*)d_in[5];

  _Float16* qkv_pack  = (_Float16*)d_ws;                              // 96 KB
  _Float16* proj_pack = (_Float16*)((char*)d_ws + (size_t)CHN * 384 * 2);

  pack_weights_kernel<<<(CHN * 384 + 255) / 256, 256, 0, stream>>>(
      qkvW, qkv_pack, CHN, 384);
  pack_weights_kernel<<<(CHN * CHN + 255) / 256, 256, 0, stream>>>(
      projW, proj_pack, CHN, CHN);

  swin_window_attn_kernel<<<4096, 256, 0, stream>>>(
      x, qkvB, projB, relT, qkv_pack, proj_pack, (float*)d_out);
}
